// Seq2Seq_inference_89215060672556
// MI455X (gfx1250) — compile-verified, matrix operand order checked by reading, not run on hardware
//
#include <hip/hip_runtime.h>
#include <hip/hip_bf16.h>
#include <math.h>

typedef __bf16 bf16;
typedef __attribute__((ext_vector_type(16))) __bf16 v16bf;
typedef __attribute__((ext_vector_type(8)))  __bf16 v8bf;
typedef __attribute__((ext_vector_type(8)))  float  v8f;

#define NBLK 64
#define TPB  128            // 4 waves of 32 (wave32)
#define NWAVES (NBLK*4)

constexpr int Hd   = 1024;
constexpr int Ed   = 1024;
constexpr int Vv   = 50257;
constexpr int Vpad = 50272;           // 3142 * 16
constexpr int NTILES = Vpad / 16;     // 3142
constexpr int Ss   = 512;
constexpr int NST  = 128;

// ---------------- workspace layout (bytes) ----------------
constexpr size_t BAR_OFF  = 0;        // 2048 * u32 barrier counters
constexpr size_t EXPS_OFF = 8192;     // 128 * f32 per-step softmax sums
constexpr size_t GMAX_OFF = 8704;     // 1 * f32 global max
constexpr size_t PMAX_OFF = 9216;     // 64 * f32 per-block max partials
constexpr size_t PARG_OFF = 9472;     // 64 * i32 per-block argmax partials
constexpr size_t HBF_OFF  = 10240;    // 2*1024 bf16 ping-pong h
constexpr size_t C_OFF    = 14336;    // 1024 f32 cell state
constexpr size_t LOG_OFF  = 18432;    // 50272 f32 logits scratch
constexpr size_t W_OFF    = 229376;   // bf16 weight region
constexpr size_t N_LSTM_W = 4096ull * 1024ull;          // elements per LSTM weight
constexpr size_t N_WOUT   = (size_t)Vpad * 1024ull;     // padded W_out elements

// ---------------- f32 -> bf16 conversion (zero-pads tail) ----------------
__global__ void k_cvt_bf16(const float* __restrict__ src, bf16* __restrict__ dst,
                           long n, long npad) {
  long i = (long)blockIdx.x * blockDim.x + threadIdx.x;
  long stride = (long)gridDim.x * blockDim.x;
  for (; i < npad; i += stride)
    dst[i] = (i < n) ? (bf16)src[i] : (bf16)0.0f;
}

// ---------------- per-launch state init (runs before main on same stream) ----
__global__ void k_init(unsigned* bars, float* expsum, unsigned short* hbf,
                       float* cbuf) {
  int i = blockIdx.x * blockDim.x + threadIdx.x;
  int stride = gridDim.x * blockDim.x;
  for (int k = i; k < 2048;   k += stride) bars[k] = 0u;
  for (int k = i; k < NST;    k += stride) expsum[k] = 0.0f;
  for (int k = i; k < 2 * Hd; k += stride) hbf[k] = 0;     // bf16 zero
  for (int k = i; k < Hd;     k += stride) cbuf[k] = 0.0f;
}

// ---------------- grid-wide barrier: one fresh counter per instance ----------
__device__ __forceinline__ void gbar(unsigned* bars, int idx) {
  __syncthreads();
  __threadfence();
  if (threadIdx.x == 0) {
    atomicAdd(&bars[idx], 1u);
    volatile unsigned* p = bars + idx;
    while (*p < (unsigned)NBLK) __builtin_amdgcn_s_sleep(1);
  }
  __syncthreads();
  __threadfence();
}

// ---------------- 16-row matvec tile via WMMA bf16 (broadcast-B) -------------
// A tile: 16 weight rows (ld = 1024), K swept in chunks of 32.
// ISA A layout (16-bit, 16x32): lanes 0-15 hold K {0..7,16..23}; lanes 16-31
// hold K {8..15,24..31}; M = lane&15. B is the bf16 vector broadcast into the
// matching K slots (N columns redundant). D: acc[i] = y[i] on lanes 0-15,
// y[8+i] on lanes 16-31.
__device__ __forceinline__ v8f wmma_mv16(const bf16* __restrict__ Wtile,
                                         const bf16* __restrict__ xv,
                                         int K, v8f acc) {
  const int lane = threadIdx.x & 31;
  const int m    = lane & 15;
  const int half = lane >> 4;
  const bf16* wrow = Wtile + (size_t)m * 1024;
#pragma unroll 4
  for (int k0 = 0; k0 < K; k0 += 32) {
    v8bf alo = *(const v8bf*)(wrow + k0 + half * 8);
    v8bf ahi = *(const v8bf*)(wrow + k0 + half * 8 + 16);
    v8bf blo = *(const v8bf*)(xv + k0 + half * 8);
    v8bf bhi = *(const v8bf*)(xv + k0 + half * 8 + 16);
    v16bf a, b;
#pragma unroll
    for (int i = 0; i < 8; ++i) {
      a[i] = alo[i]; a[8 + i] = ahi[i];
      b[i] = blo[i]; b[8 + i] = bhi[i];
    }
    acc = __builtin_amdgcn_wmma_f32_16x16x32_bf16(false, a, false, b,
                                                  (short)0, acc, false, false);
  }
  return acc;
}

__device__ __forceinline__ float sigmoidf_(float x) {
  return 1.0f / (1.0f + __expf(-x));
}

// ---------------- one LSTM step: block owns h[16*b .. 16*b+16) ---------------
__device__ __forceinline__ void lstm_step(const float* __restrict__ xv,
                                          const bf16* __restrict__ hprev_bf,
                                          const bf16* __restrict__ Wih,
                                          const bf16* __restrict__ Whh,
                                          const float* __restrict__ bih,
                                          const float* __restrict__ bhh,
                                          bf16* __restrict__ hnext_bf,
                                          float* __restrict__ cvec,
                                          float* glds /* shared [64] */,
                                          bf16* xlds /* shared [1024] */) {
  // convert this step's input vector to bf16 once per block (LDS)
  for (int i = threadIdx.x; i < Ed; i += TPB) xlds[i] = (bf16)xv[i];
  __syncthreads();

  const int w    = threadIdx.x >> 5;       // gate index 0..3 (i,f,g,o)
  const int lane = threadIdx.x & 31;
  const int row0 = w * 1024 + (int)blockIdx.x * 16;

  v8f acc;
#pragma unroll
  for (int i = 0; i < 8; ++i) acc[i] = 0.0f;
  acc = wmma_mv16(Wih + (size_t)row0 * 1024, xlds,     Ed, acc);
  acc = wmma_mv16(Whh + (size_t)row0 * 1024, hprev_bf, Hd, acc);

  if (lane == 0) {
#pragma unroll
    for (int i = 0; i < 8; ++i)
      glds[w * 16 + i] = acc[i] + bih[row0 + i] + bhh[row0 + i];
  } else if (lane == 16) {
#pragma unroll
    for (int i = 0; i < 8; ++i)
      glds[w * 16 + 8 + i] = acc[i] + bih[row0 + 8 + i] + bhh[row0 + 8 + i];
  }
  __syncthreads();
  if (threadIdx.x < 16) {
    int j = (int)blockIdx.x * 16 + threadIdx.x;
    float ig = sigmoidf_(glds[0 * 16 + threadIdx.x]);
    float fg = sigmoidf_(glds[1 * 16 + threadIdx.x]);
    float gg = tanhf(glds[2 * 16 + threadIdx.x]);
    float og = sigmoidf_(glds[3 * 16 + threadIdx.x]);
    float c  = fg * cvec[j] + ig * gg;
    cvec[j]  = c;
    hnext_bf[j] = (bf16)(og * tanhf(c));
  }
  __syncthreads();
}

// ---------------- persistent main kernel ------------------------------------
__global__ __launch_bounds__(TPB) void k_main(
    const int* __restrict__ source, const int* __restrict__ start_tok,
    const float* __restrict__ Wemb_enc, const float* __restrict__ bih_e,
    const float* __restrict__ bhh_e,
    const float* __restrict__ Wemb_dec, const float* __restrict__ bih_d,
    const float* __restrict__ bhh_d,
    const float* __restrict__ b_out,
    const bf16* __restrict__ WihE, const bf16* __restrict__ WhhE,
    const bf16* __restrict__ WihD, const bf16* __restrict__ WhhD,
    const bf16* __restrict__ Wout,
    bf16* hbf, float* cbuf, float* logits, float* expsum, float* gmax,
    float* pmax, int* parg, unsigned* bars, int* toks, float* probs) {
  __shared__ __align__(16) bf16 xlds[Ed];
  __shared__ float glds[64];
  __shared__ float wmaxs[8];
  __shared__ int   wargs[8];
  __shared__ float smax[64];
  __shared__ int   sidx[64];

  int bar = 0;
  const int lane  = threadIdx.x & 31;
  const int wave  = threadIdx.x >> 5;
  const int gwave = (int)blockIdx.x * 4 + wave;
  const int gtid  = (int)blockIdx.x * TPB + threadIdx.x;

  // ---------------- encoder: 512 sequential steps ----------------
  for (int t = 0; t < Ss; ++t) {
    const float* xv = Wemb_enc + (size_t)source[t] * Ed;
    const bf16*  hp = hbf + (size_t)(t & 1) * Hd;
    bf16*        hn = hbf + (size_t)((t + 1) & 1) * Hd;
    lstm_step(xv, hp, WihE, WhhE, bih_e, bhh_e, hn, cbuf, glds, xlds);
    gbar(bars, bar++);
  }

  // ---------------- decoder: 128 greedy steps ----------------
  for (int s = 0; s < NST; ++s) {
    const int T   = Ss + s;
    const int tok = (s == 0) ? start_tok[0] : toks[s - 1];
    const float* xv = Wemb_dec + (size_t)tok * Ed;
    const bf16*  hp = hbf + (size_t)(T & 1) * Hd;
    bf16*        hn = hbf + (size_t)((T + 1) & 1) * Hd;
    lstm_step(xv, hp, WihD, WhhD, bih_d, bhh_d, hn, cbuf, glds, xlds);
    gbar(bars, bar++);                                  // b1: h ready

    // logits = W_out @ h + b_out (bf16 W_out resident in 192MB L2)
    // fold running (max, first-index) into the write loop
    float lm = -3.402823466e38f; int li = 0x7fffffff;
    for (int tile = gwave; tile < NTILES; tile += NWAVES) {
      const int r0 = tile * 16;
      v8f acc;
#pragma unroll
      for (int i = 0; i < 8; ++i) acc[i] = 0.0f;
      if (r0 + NWAVES * 16 < Vpad)
        __builtin_prefetch(Wout + (size_t)(r0 + NWAVES * 16) * 1024, 0, 1);
      acc = wmma_mv16(Wout + (size_t)r0 * 1024, hn, Hd, acc);
      if (lane == 0) {
#pragma unroll
        for (int i = 0; i < 8; ++i) {
          int r = r0 + i;
          if (r < Vv) {
            float val = acc[i] + b_out[r];
            logits[r] = val;
            if (val > lm) { lm = val; li = r; }
          } else {
            logits[r] = -3.402823466e38f;
          }
        }
      } else if (lane == 16) {
#pragma unroll
        for (int i = 0; i < 8; ++i) {
          int r = r0 + 8 + i;
          if (r < Vv) {
            float val = acc[i] + b_out[r];
            logits[r] = val;
            if (val > lm) { lm = val; li = r; }
          } else {
            logits[r] = -3.402823466e38f;
          }
        }
      }
    }
    if (lane == 0)       { wmaxs[wave * 2]     = lm; wargs[wave * 2]     = li; }
    else if (lane == 16) { wmaxs[wave * 2 + 1] = lm; wargs[wave * 2 + 1] = li; }
    __syncthreads();
    if (threadIdx.x == 0) {
      float bm = wmaxs[0]; int bi = wargs[0];
#pragma unroll
      for (int k = 1; k < 8; ++k) {
        float ov = wmaxs[k]; int oi = wargs[k];
        if (ov > bm || (ov == bm && oi < bi)) { bm = ov; bi = oi; }
      }
      pmax[blockIdx.x] = bm; parg[blockIdx.x] = bi;
    }
    gbar(bars, bar++);                                  // b2: logits+partials

    // block 0 reduces the 64 per-block partials
    if (blockIdx.x == 0) {
      if (threadIdx.x < 64) { smax[threadIdx.x] = pmax[threadIdx.x];
                              sidx[threadIdx.x] = parg[threadIdx.x]; }
      __syncthreads();
      for (int off = 32; off > 0; off >>= 1) {
        if (threadIdx.x < off) {
          float ov = smax[threadIdx.x + off]; int oi = sidx[threadIdx.x + off];
          if (ov > smax[threadIdx.x] ||
              (ov == smax[threadIdx.x] && oi < sidx[threadIdx.x])) {
            smax[threadIdx.x] = ov; sidx[threadIdx.x] = oi;
          }
        }
        __syncthreads();
      }
      if (threadIdx.x == 0) { gmax[0] = smax[0]; toks[s] = sidx[0]; }
    }
    gbar(bars, bar++);                                  // b3: token + max ready

    // softmax pass 1: exp + per-step sum
    const float gm = gmax[0];
    float* pout = probs + (size_t)s * Vv;
    float lsum = 0.0f;
    for (int v = gtid; v < Vv; v += NBLK * TPB) {
      float e = __expf(logits[v] - gm);
      pout[v] = e;
      lsum += e;
    }
    atomicAdd(&expsum[s], lsum);
    gbar(bars, bar++);                                  // b4: sum ready

    // softmax pass 2: normalize
    const float inv = 1.0f / expsum[s];
    for (int v = gtid; v < Vv; v += NBLK * TPB) pout[v] *= inv;
  }
}

extern "C" void kernel_launch(void* const* d_in, const int* in_sizes, int n_in,
                              void* d_out, int out_size, void* d_ws, size_t ws_size,
                              hipStream_t stream) {
  const int*   source   = (const int*)  d_in[0];
  const int*   start_tk = (const int*)  d_in[2];
  const float* Wemb_enc = (const float*)d_in[3];
  const float* W_ih_e   = (const float*)d_in[4];
  const float* W_hh_e   = (const float*)d_in[5];
  const float* b_ih_e   = (const float*)d_in[6];
  const float* b_hh_e   = (const float*)d_in[7];
  const float* Wemb_dec = (const float*)d_in[8];
  const float* W_ih_d   = (const float*)d_in[9];
  const float* W_hh_d   = (const float*)d_in[10];
  const float* b_ih_d   = (const float*)d_in[11];
  const float* b_hh_d   = (const float*)d_in[12];
  const float* W_out    = (const float*)d_in[13];
  const float* b_out    = (const float*)d_in[14];

  char* ws = (char*)d_ws;
  unsigned* bars   = (unsigned*)(ws + BAR_OFF);
  float*    expsum = (float*)   (ws + EXPS_OFF);
  float*    gmax   = (float*)   (ws + GMAX_OFF);
  float*    pmax   = (float*)   (ws + PMAX_OFF);
  int*      parg   = (int*)     (ws + PARG_OFF);
  bf16*     hbf    = (bf16*)    (ws + HBF_OFF);
  float*    cbuf   = (float*)   (ws + C_OFF);
  float*    logits = (float*)   (ws + LOG_OFF);
  bf16*     wb     = (bf16*)    (ws + W_OFF);
  bf16* WihE = wb;
  bf16* WhhE = wb + N_LSTM_W;
  bf16* WihD = wb + 2 * N_LSTM_W;
  bf16* WhhD = wb + 3 * N_LSTM_W;
  bf16* Wout = wb + 4 * N_LSTM_W;

  int*   toks  = (int*)d_out;
  float* probs = (float*)d_out + NST;

  // per-launch weight conversion f32 -> bf16 (makes all weights L2-resident)
  k_cvt_bf16<<<2048, 256, 0, stream>>>(W_ih_e, WihE, (long)N_LSTM_W, (long)N_LSTM_W);
  k_cvt_bf16<<<2048, 256, 0, stream>>>(W_hh_e, WhhE, (long)N_LSTM_W, (long)N_LSTM_W);
  k_cvt_bf16<<<2048, 256, 0, stream>>>(W_ih_d, WihD, (long)N_LSTM_W, (long)N_LSTM_W);
  k_cvt_bf16<<<2048, 256, 0, stream>>>(W_hh_d, WhhD, (long)N_LSTM_W, (long)N_LSTM_W);
  k_cvt_bf16<<<4096, 256, 0, stream>>>(W_out,  Wout, (long)Vv * 1024L, (long)N_WOUT);

  // reset all persistent-kernel state (stream-ordered before k_main)
  k_init<<<16, 256, 0, stream>>>(bars, expsum, (unsigned short*)hbf, cbuf);

  k_main<<<NBLK, TPB, 0, stream>>>(
      source, start_tk, Wemb_enc, b_ih_e, b_hh_e, Wemb_dec, b_ih_d, b_hh_d,
      b_out, WihE, WhhE, WihD, WhhD, Wout,
      hbf, cbuf, logits, expsum, gmax, pmax, parg, bars, toks, probs);
}
